// learnSystemMLP_47571057771080
// MI455X (gfx1250) — matrix operand order repaired and checked by reading, not compiled, not run on hardware
//
#include <hip/hip_runtime.h>
#include <hip/hip_bf16.h>

// ---------------------------------------------------------------------------
// CDNA5 (gfx1250) wave32 WMMA bf16x3 fused kernels for learnSystemMLP.
// ---------------------------------------------------------------------------

typedef __attribute__((ext_vector_type(16))) __bf16 v16bf;
typedef __attribute__((ext_vector_type(8)))  float  v8f;

__device__ __forceinline__ unsigned short f2bf(float f) {
  // round-to-nearest-even float -> bf16
  unsigned u = __builtin_bit_cast(unsigned, f);
  u += 0x7FFFu + ((u >> 16) & 1u);
  return (unsigned short)(u >> 16);
}
__device__ __forceinline__ float bf2f(unsigned short h) {
  unsigned u = ((unsigned)h) << 16;
  return __builtin_bit_cast(float, u);
}
__device__ __forceinline__ float silu_f(float z) {
  return z / (1.0f + __expf(-z));
}

union FragU { v16bf v; unsigned u[8]; };

// A-matrix fragment: 16(M) x 32(K) bf16 tile from LDS, row-major [16][ldk].
// ISA 7.12.2 layout: lanes 0-15 -> M=lane, K in {0..7,16..23}; lanes 16-31
// hold K in {8..15,24..31}; VGPR v holds K pair (2 bf16 = 1 dword).
__device__ __forceinline__ v16bf load_a_frag(const unsigned short* x, int ldk,
                                             int k0, int lane) {
  int m  = lane & 15;
  int kb = (lane >> 4) << 3;              // 0 or 8
  const unsigned short* row = x + m * ldk + k0 + kb;
  FragU f;
#pragma unroll
  for (int v = 0; v < 8; ++v) {
    int k = (v < 4) ? (2 * v) : (16 + 2 * (v - 4));
    f.u[v] = *(const unsigned*)(row + k);
  }
  return f.v;
}

// B-matrix fragment: 32(K) x 16(N) bf16 tile, weights stored [N][K] in global
// so K pairs are contiguous. Lanes 0-15 -> N=lane, K {0..7,16..23}; lanes 16-31 K {8..15,24..31}.
__device__ __forceinline__ v16bf load_b_frag(const unsigned short* w, int K,
                                             int col0, int k0, int lane) {
  int n  = lane & 15;
  int kb = (lane >> 4) << 3;              // 0 or 8
  const unsigned short* p = w + (size_t)(col0 + n) * K + k0 + kb;
  FragU f;
#pragma unroll
  for (int v = 0; v < 8; ++v) f.u[v] = *(const unsigned*)(p + ((v < 4) ? (2 * v) : (16 + 2 * (v - 4))));
  return f.v;
}

__device__ __forceinline__ v8f wmma_bf16(v16bf a, v16bf b, v8f c) {
  // (neg_a, A, neg_b, B, c_mod, C, reuse_a, reuse_b)
  return __builtin_amdgcn_wmma_f32_16x16x32_bf16(false, a, false, b,
                                                 (short)0, c, false, false);
}

// ---------------------------------------------------------------------------
// Split f32 weights into hi/lo bf16, transposed to [N][K] for frag loads.
// ---------------------------------------------------------------------------
__global__ __launch_bounds__(256) void k_convert(const float* __restrict__ src,
                                                 unsigned short* __restrict__ hi,
                                                 unsigned short* __restrict__ lo,
                                                 int K, int N) {
  int idx = blockIdx.x * 256 + threadIdx.x;
  if (idx >= K * N) return;
  int k = idx / N;
  int n = idx - k * N;
  float a = src[idx];
  unsigned short h = f2bf(a);
  hi[(size_t)n * K + k] = h;
  lo[(size_t)n * K + k] = f2bf(a - bf2f(h));
}

// ---------------------------------------------------------------------------
// H kernel: 16-row strip per block.  M = silu(x@A_H) (16x800) computed by
// WMMA, squared & group-reduced into macc[5][16][32] (m11,m12,m21,m22,mpp),
// then the quadratic form with q built from x gives H per row.
// ---------------------------------------------------------------------------
__global__ __launch_bounds__(256) void k_H(const float* __restrict__ xh,
                                           const unsigned short* __restrict__ Whi,
                                           const unsigned short* __restrict__ Wlo,
                                           float* __restrict__ outH) {
  __shared__ float sx[16 * 192];
  __shared__ unsigned short sxh[16 * 192], sxl[16 * 192];
  __shared__ float macc[5][16][32];
  __shared__ float part[256];
  const int tid = threadIdx.x, wave = tid >> 5, lane = tid & 31;
  const int row0 = blockIdx.x * 16;

  const float* xr = xh + (size_t)row0 * 192;
  for (int i = tid; i < 16 * 192; i += 256) {
    float v = xr[i];
    sx[i] = v;
    unsigned short h = f2bf(v);
    sxh[i] = h;
    sxl[i] = f2bf(v - bf2f(h));
  }
  for (int i = tid; i < 5 * 16 * 32; i += 256) (&macc[0][0][0])[i] = 0.f;
  __syncthreads();

  for (int t = wave; t < 50; t += 8) {          // 800/16 = 50 N-tiles
    v8f acc = {};
#pragma unroll
    for (int ks = 0; ks < 6; ++ks) {            // K = 192 = 6 * 32
      v16bf ah = load_a_frag(sxh, 192, ks * 32, lane);
      v16bf al = load_a_frag(sxl, 192, ks * 32, lane);
      v16bf bh = load_b_frag(Whi, 192, t * 16, ks * 32, lane);
      v16bf bl = load_b_frag(Wlo, 192, t * 16, ks * 32, lane);
      acc = wmma_bf16(al, bh, acc);             // x_lo * A_hi
      acc = wmma_bf16(ah, bl, acc);             // x_hi * A_lo
      acc = wmma_bf16(ah, bh, acc);             // x_hi * A_hi
    }
    int col = t * 16 + (lane & 15);
    int g = col >> 5, a = col & 31;
    int tgt = (g < 6) ? 0 : (g < 12) ? 1 : (g < 18) ? 2 : (g < 24) ? 3 : 4;
#pragma unroll
    for (int r = 0; r < 8; ++r) {               // C/D: row = r + 8*(lane>=16)
      int m = r + ((lane >> 4) << 3);
      float s = silu_f(acc[r]);
      atomicAdd(&macc[tgt][m][a], s * s);
    }
  }
  __syncthreads();

  {
    int m = tid >> 4, l16 = tid & 15;
    float h = 0.f;
#pragma unroll
    for (int jj = 0; jj < 4; ++jj) {
      int j = l16 + 16 * jj;                    // j in 0..63
      int a = j >> 2, c = j & 3, t2 = j >> 1;
      float qa = sx[m * 192 + 6 * a + c];
      float qb = sx[m * 192 + 6 * (a + 16) + c];
      h += qa * qa * macc[0][m][t2] + qa * qb * (macc[1][m][t2] + macc[2][m][t2]) +
           qb * qb * macc[3][m][t2];
    }
    h += macc[4][m][l16] + macc[4][m][l16 + 16];  // mpp
    part[tid] = h;
  }
  __syncthreads();
  if (tid < 16) {
    float s = 0.f;
    for (int i = 0; i < 16; ++i) s += part[tid * 16 + i];
    outH[row0 + tid] = s;
  }
}

// ---------------------------------------------------------------------------
// J GEMM kernel: j[b] = sum_c silu((x_jr@A_J)[b,c]); 16-row strip per block.
// ---------------------------------------------------------------------------
__global__ __launch_bounds__(256) void k_Jgemm(const float* __restrict__ xjr,
                                               const unsigned short* __restrict__ Whi,
                                               const unsigned short* __restrict__ Wlo,
                                               float* __restrict__ jws) {
  __shared__ unsigned short sxh[16 * 128], sxl[16 * 128];
  __shared__ float jacc[16];
  const int tid = threadIdx.x, wave = tid >> 5, lane = tid & 31;
  const int row0 = blockIdx.x * 16;

  const float* xr = xjr + (size_t)row0 * 128;
  for (int i = tid; i < 16 * 128; i += 256) {
    float v = xr[i];
    unsigned short h = f2bf(v);
    sxh[i] = h;
    sxl[i] = f2bf(v - bf2f(h));
  }
  if (tid < 16) jacc[tid] = 0.f;
  __syncthreads();

  float rowpart[8];
#pragma unroll
  for (int r = 0; r < 8; ++r) rowpart[r] = 0.f;

  for (int t = wave; t < 32; t += 8) {          // 512/16 = 32 N-tiles
    v8f acc = {};
#pragma unroll
    for (int ks = 0; ks < 4; ++ks) {            // K = 128
      v16bf ah = load_a_frag(sxh, 128, ks * 32, lane);
      v16bf al = load_a_frag(sxl, 128, ks * 32, lane);
      v16bf bh = load_b_frag(Whi, 128, t * 16, ks * 32, lane);
      v16bf bl = load_b_frag(Wlo, 128, t * 16, ks * 32, lane);
      acc = wmma_bf16(al, bh, acc);
      acc = wmma_bf16(ah, bl, acc);
      acc = wmma_bf16(ah, bh, acc);
    }
#pragma unroll
    for (int r = 0; r < 8; ++r) rowpart[r] += silu_f(acc[r]);
  }
  // reduce across the 16 lanes that share a row (xor stays in the half-wave)
#pragma unroll
  for (int r = 0; r < 8; ++r) {
    float v = rowpart[r];
    for (int off = 1; off < 16; off <<= 1) v += __shfl_xor(v, off, 32);
    if ((lane & 15) == 0) atomicAdd(&jacc[r + ((lane >> 4) << 3)], v);
  }
  __syncthreads();
  if (tid < 16) jws[row0 + tid] = jacc[tid];
}

// ---------------------------------------------------------------------------
// J writer: one thread per 128-float row; zeros + single scattered element.
// J[b0][u][64+u] = -d, J[b0][64+u][u] = +d, d = j[b0, (u&63)>>1].
// ---------------------------------------------------------------------------
__global__ __launch_bounds__(256) void k_Jwrite(const float* __restrict__ jws,
                                                float* __restrict__ outJ) {
  int row = blockIdx.x * 256 + threadIdx.x;     // 1024*128 rows
  int b0 = row >> 7;
  int u  = row & 127;
  float d = jws[b0 * 32 + ((u & 63) >> 1)];
  int   nz_col = (u < 64) ? (64 + u) : (u - 64);
  float nz_val = (u < 64) ? -d : d;
  float4* dst = (float4*)(outJ + (size_t)row * 128);
#pragma unroll 4
  for (int c4 = 0; c4 < 32; ++c4) {
    float4 w = make_float4(0.f, 0.f, 0.f, 0.f);
    if (c4 == (nz_col >> 2)) ((float*)&w)[nz_col & 3] = nz_val;
    dst[c4] = w;
  }
}

// ---------------------------------------------------------------------------
// R kernel: one batch b0 per block (32 rows x 512 cols GEMM), accumulate the
// four 32x32 blocks r11,r12,r21,r22, then emit the 128x128 Laplacian-like
// output:  diag(u) = full row sum of Ru,  offdiag = -Ru[u,v],
//          Ru[u,v] = S[u,v] + S[v,u],  S = Rfull^2 (parity-structured).
// ---------------------------------------------------------------------------
__global__ __launch_bounds__(256) void k_R(const float* __restrict__ xjr,
                                           const unsigned short* __restrict__ Whi,
                                           const unsigned short* __restrict__ Wlo,
                                           float* __restrict__ outR) {
  __shared__ unsigned short sxh[32 * 128], sxl[32 * 128];
  __shared__ float racc[4][32][32];
  __shared__ float rowsq[4][32], colsq[4][32];
  const int tid = threadIdx.x, wave = tid >> 5, lane = tid & 31;
  const int b0 = blockIdx.x;

  const float* xr = xjr + (size_t)b0 * 32 * 128;
  for (int i = tid; i < 32 * 128; i += 256) {
    float v = xr[i];
    unsigned short h = f2bf(v);
    sxh[i] = h;
    sxl[i] = f2bf(v - bf2f(h));
  }
  for (int i = tid; i < 4 * 32 * 32; i += 256) (&racc[0][0][0])[i] = 0.f;
  __syncthreads();

  for (int t = wave; t < 64; t += 8) {          // 2 row-tiles x 32 col-tiles
    int rt = t >> 5, ct = t & 31;
    const unsigned short* axh = sxh + rt * 16 * 128;
    const unsigned short* axl = sxl + rt * 16 * 128;
    v8f acc = {};
#pragma unroll
    for (int ks = 0; ks < 4; ++ks) {
      v16bf ah = load_a_frag(axh, 128, ks * 32, lane);
      v16bf al = load_a_frag(axl, 128, ks * 32, lane);
      v16bf bh = load_b_frag(Whi, 128, ct * 16, ks * 32, lane);
      v16bf bl = load_b_frag(Wlo, 128, ct * 16, ks * 32, lane);
      acc = wmma_bf16(al, bh, acc);
      acc = wmma_bf16(ah, bl, acc);
      acc = wmma_bf16(ah, bh, acc);
    }
    int col = ct * 16 + (lane & 15);
    int s = col >> 7;                           // block index (g>>2)
    int ap = col & 31;
#pragma unroll
    for (int r = 0; r < 8; ++r) {
      int a = rt * 16 + r + ((lane >> 4) << 3); // row atom within batch
      atomicAdd(&racc[s][a][ap], silu_f(acc[r]));
    }
  }
  __syncthreads();

  if (tid < 128) {
    int s = tid >> 5, i = tid & 31;
    float acc = 0.f;
    for (int j = 0; j < 32; ++j) { float v = racc[s][i][j]; acc += v * v; }
    rowsq[s][i] = acc;
  } else {
    int s = (tid - 128) >> 5, j = (tid - 128) & 31;
    float acc = 0.f;
    for (int i = 0; i < 32; ++i) { float v = racc[s][i][j]; acc += v * v; }
    colsq[s][j] = acc;
  }
  __syncthreads();

  float4* dst = (float4*)(outR + (size_t)b0 * 16384);
  for (int e4 = tid; e4 < 4096; e4 += 256) {
    int u  = e4 >> 5;
    int v0 = (e4 & 31) << 2;
    int uh = u >> 6, iu = (u & 63) >> 1;
    float4 w;
    float* wp = (float*)&w;
#pragma unroll
    for (int q = 0; q < 4; ++q) {
      int v = v0 + q;
      float rres;
      if (v == u) {
        rres = (uh == 0)
                 ? (rowsq[0][iu] + rowsq[1][iu] + colsq[0][iu] + colsq[2][iu])
                 : (rowsq[2][iu] + rowsq[3][iu] + colsq[1][iu] + colsq[3][iu]);
      } else if (((u ^ v) & 1) == 0) {
        int vh = v >> 6, jv = (v & 63) >> 1;
        float a1 = racc[uh * 2 + vh][iu][jv];   // S[u,v]
        float a2 = racc[vh * 2 + uh][jv][iu];   // S[v,u]
        rres = -(a1 * a1 + a2 * a2);
      } else {
        rres = 0.f;
      }
      wp[q] = rres;
    }
    dst[e4] = w;
  }
}

// ---------------------------------------------------------------------------
extern "C" void kernel_launch(void* const* d_in, const int* in_sizes, int n_in,
                              void* d_out, int out_size, void* d_ws, size_t ws_size,
                              hipStream_t stream) {
  (void)in_sizes; (void)n_in; (void)out_size; (void)ws_size;
  const float* x_h  = (const float*)d_in[0];   // (32768, 192)
  const float* x_jr = (const float*)d_in[1];   // (32768, 128)
  const float* A_H  = (const float*)d_in[2];   // (192, 800)
  const float* A_J  = (const float*)d_in[3];   // (128, 512)
  const float* A_R  = (const float*)d_in[4];   // (128, 512)

  // workspace: bf16 hi/lo weights (transposed [N][K]) + j row sums
  unsigned short* AHhi = (unsigned short*)d_ws;
  unsigned short* AHlo = AHhi + 153600;        // 192*800
  unsigned short* AJhi = AHlo + 153600;
  unsigned short* AJlo = AJhi + 65536;         // 128*512
  unsigned short* ARhi = AJlo + 65536;
  unsigned short* ARlo = ARhi + 65536;
  float* jws = (float*)(ARlo + 65536);         // 32768 floats

  float* outH = (float*)d_out;                 // 32768
  float* outJ = outH + 32768;                  // 1024*128*128
  float* outR = outJ + (size_t)1024 * 128 * 128;

  hipLaunchKernelGGL(k_convert, dim3(600), dim3(256), 0, stream, A_H, AHhi, AHlo, 192, 800);
  hipLaunchKernelGGL(k_convert, dim3(256), dim3(256), 0, stream, A_J, AJhi, AJlo, 128, 512);
  hipLaunchKernelGGL(k_convert, dim3(256), dim3(256), 0, stream, A_R, ARhi, ARlo, 128, 512);

  hipLaunchKernelGGL(k_H,     dim3(2048), dim3(256), 0, stream, x_h,  AHhi, AHlo, outH);
  hipLaunchKernelGGL(k_Jgemm, dim3(2048), dim3(256), 0, stream, x_jr, AJhi, AJlo, jws);
  hipLaunchKernelGGL(k_Jwrite, dim3(512), dim3(256), 0, stream, jws, outJ);
  hipLaunchKernelGGL(k_R,     dim3(1024), dim3(256), 0, stream, x_jr, ARhi, ARlo, outR);
}
